// PrototypeLayer_70574902607955
// MI455X (gfx1250) — matrix operand order repaired and checked by reading, not compiled, not run on hardware
//
#include <hip/hip_runtime.h>

typedef __bf16 bf16_t;
typedef __attribute__((ext_vector_type(2)))  __bf16 v2bf;
typedef __attribute__((ext_vector_type(8)))  __bf16 v8bf;
typedef __attribute__((ext_vector_type(16))) __bf16 v16bf;
typedef __attribute__((ext_vector_type(8)))  float  v8f;
typedef __attribute__((ext_vector_type(4)))  unsigned int u32x4;
typedef __attribute__((ext_vector_type(4)))  int i32x4;
typedef __attribute__((ext_vector_type(8)))  int i32x8;

#define CAT16(a,b) __builtin_shufflevector((a),(b),0,1,2,3,4,5,6,7,8,9,10,11,12,13,14,15)

namespace {
constexpr int kB   = 32;     // batch
constexpr int kC   = 512;    // channels (K dim)
constexpr int kHW  = 784;    // 28*28 (M dim, 49 tiles of 16)
constexpr int kP   = 2000;   // prototypes (N dim)
constexpr int kNT  = 64;     // prototypes per workgroup (4 waves x 16)
constexpr int kSK  = 520;    // padded K stride in halves (1040B row -> 4-bank shift, conflict-free)

// ---- TDM-path LDS layout: A buffers LOW so in-loop DS offsets fit the 16-bit imm ----
constexpr unsigned kOffA  = 0;                       // 4 A arrays: buf0{hi,lo}, buf1{hi,lo}
constexpr unsigned kABuf  = 16 * kSK * 2;            // 16,640 B each
constexpr unsigned kOffBH = kOffA + 4 * kABuf;       // B hi [64][520] halves @ 66,560
constexpr unsigned kOffBL = kOffBH + kNT * kSK * 2;  // B lo @ 133,120
constexpr unsigned kSmemTdm = kOffBL + kNT * kSK * 2;  // 199,680 B (< 320 KB/WG)

// ---- fallback-path LDS ----
constexpr int kKC  = 256;    // K chunk for in-kernel A staging
constexpr int kSKA = 264;
constexpr int kSmemFb = (2 * kNT * kSK + 2 * 16 * kSKA) * 2;  // 150,016 B

// ---- workspace layout (bytes) ----
constexpr size_t kOffX2 = 0;                               // 32*784 f32
constexpr size_t kOffP2 = 102400;                          // 2000 f32
constexpr size_t kOffXh = 114688;                          // [32][784][512] bf16 (transposed hi)
constexpr size_t kOffXl = kOffXh + (size_t)kB*kHW*kC*2;    // lo
constexpr size_t kOffPh = kOffXl + (size_t)kB*kHW*kC*2;    // [2000][512] bf16 hi
constexpr size_t kOffPl = kOffPh + (size_t)kP*kC*2;        // lo
constexpr size_t kWsTdm = kOffPl + (size_t)kP*kC*2;        // ~53 MB total
}

// ============================ norm precompute ============================
__global__ __launch_bounds__(256) void x2_kernel(const float* __restrict__ feat,
                                                 float* __restrict__ x2w) {
  int b  = blockIdx.x >> 2;
  int hw = (blockIdx.x & 3) * 256 + threadIdx.x;
  if (hw >= kHW) return;
  const float* p = feat + (size_t)b * kC * kHW + hw;
  float s = 0.f;
  for (int c = 0; c < kC; ++c) { float v = p[(size_t)c * kHW]; s = fmaf(v, v, s); }
  x2w[b * kHW + hw] = s;
}

__global__ __launch_bounds__(32) void p2_kernel(const float* __restrict__ prot,
                                                float* __restrict__ p2w) {
  int n = blockIdx.x, t = threadIdx.x;
  const float4* p = (const float4*)(prot + (size_t)n * kC) + t * 4;
  float s = 0.f;
#pragma unroll
  for (int i = 0; i < 4; ++i) {
    float4 v = p[i];
    s = fmaf(v.x, v.x, s); s = fmaf(v.y, v.y, s);
    s = fmaf(v.z, v.z, s); s = fmaf(v.w, v.w, s);
  }
#pragma unroll
  for (int off = 16; off > 0; off >>= 1) s += __shfl_xor(s, off, 32);
  if (t == 0) p2w[n] = s;
}

// ================== bf16 hi/lo precompute (TDM path) =====================
__global__ __launch_bounds__(256) void cvt_feat(const float* __restrict__ feat,
                                                bf16_t* __restrict__ Xh,
                                                bf16_t* __restrict__ Xl) {
  __shared__ float tile[32][33];
  const int b = blockIdx.z, hw0 = blockIdx.x * 32, c0 = blockIdx.y * 32;
  const int x = threadIdx.x & 31, y = threadIdx.x >> 5;   // y = 0..7
#pragma unroll
  for (int i = 0; i < 4; ++i) {
    int c = c0 + y + 8 * i, hw = hw0 + x;
    tile[y + 8 * i][x] = (hw < kHW) ? feat[(size_t)b * kC * kHW + (size_t)c * kHW + hw] : 0.f;
  }
  __syncthreads();
#pragma unroll
  for (int i = 0; i < 4; ++i) {
    int hw = hw0 + y + 8 * i;
    if (hw < kHW) {
      float v = tile[x][y + 8 * i];
      bf16_t h = (bf16_t)v;
      size_t o = ((size_t)b * kHW + hw) * kC + c0 + x;
      Xh[o] = h;
      Xl[o] = (bf16_t)(v - (float)h);
    }
  }
}

__global__ __launch_bounds__(256) void cvt_prot(const float* __restrict__ prot,
                                                bf16_t* __restrict__ Ph,
                                                bf16_t* __restrict__ Pl) {
  int i = blockIdx.x * 256 + threadIdx.x;   // kP*kC = 1,024,000 elements
  if (i >= kP * kC) return;
  float v = prot[i];
  bf16_t h = (bf16_t)v;
  Ph[i] = h;
  Pl[i] = (bf16_t)(v - (float)h);
}

// ===================== TDM descriptor + issue helper =====================
// 2D row-major bf16 tile -> LDS, padded 4 DWORDs every 256 DWORDs (512-half rows -> 520-half stride).
__device__ __forceinline__ void tdm_load_2d(const bf16_t* gptr, unsigned ldsOff,
                                            unsigned tileRows, unsigned tensorRows) {
  unsigned long long ga = (unsigned long long)(uintptr_t)gptr;
  u32x4 g0;
  g0[0] = 1u;                                   // count=1, is_restore=0, no gather
  g0[1] = ldsOff;                               // lds_addr (bytes)
  g0[2] = (unsigned)ga;                         // global_addr[31:0]
  g0[3] = (unsigned)(ga >> 32) | (2u << 30);    // global_addr[56:32] | type=2 ("image")
  i32x8 g1;
  g1[0] = (1 << 16) | (1 << 20) | (7 << 22) | (3 << 25); // data_size=2B, pad_en, every 256dw, pad 4dw
  g1[1] = (int)((kC & 0xFFFF) << 16);                    // tensor_dim0 = 512 (low 16)
  g1[2] = (int)((kC >> 16) | ((tensorRows & 0xFFFF) << 16)); // dim0 hi | tensor_dim1 lo
  g1[3] = (int)((tensorRows >> 16) | (kC << 16));        // tensor_dim1 hi | tile_dim0 = 512
  g1[4] = (int)tileRows;                                 // tile_dim1 | tile_dim2=0
  g1[5] = kC;                                            // tensor_dim0_stride = 512
  g1[6] = 0; g1[7] = 0;                                  // stride hi, dim1_stride (unused, 2D)
  i32x4 z4 = {0, 0, 0, 0};
#if __clang_major__ >= 23
  i32x8 z8 = {0, 0, 0, 0, 0, 0, 0, 0};
  __builtin_amdgcn_tensor_load_to_lds(g0, g1, z4, z4, z8, 0);
#else
  __builtin_amdgcn_tensor_load_to_lds(g0, g1, z4, z4, 0);
#endif
}

// ============= main kernel, TDM path: pure DMA staging + WMMA =============
__global__ __launch_bounds__(128) void proto_tdm(const bf16_t* __restrict__ Xh,
                                                 const bf16_t* __restrict__ Xl,
                                                 const bf16_t* __restrict__ Ph,
                                                 const bf16_t* __restrict__ Pl,
                                                 const float* __restrict__ x2w,
                                                 const float* __restrict__ p2w,
                                                 float* __restrict__ out) {
  extern __shared__ char smem[];
  const int tid  = threadIdx.x;
  const int lane = tid & 31;
  const int wave = tid >> 5;
  const int ml   = lane & 15;
  const int hh   = lane >> 4;
  const int b    = blockIdx.x >> 5;
  const int n0   = (blockIdx.x & 31) * kNT;
  const int gN   = n0 + wave * 16 + ml;
  const float p2v = (gN < kP) ? p2w[gN] : 0.f;
  const unsigned bRows = (unsigned)((kP - n0 < kNT) ? (kP - n0) : kNT); // OOB rows read as 0
  const bf16_t* XhB = Xh + (size_t)b * kHW * kC;
  const bf16_t* XlB = Xl + (size_t)b * kHW * kC;

  // Prologue: B tile (once) + A tile 0 into buffer 0. One wave drives the TDM.
  if (wave == 0) {
    tdm_load_2d(Ph + (size_t)n0 * kC, kOffBH, kNT, bRows);
    tdm_load_2d(Pl + (size_t)n0 * kC, kOffBL, kNT, bRows);
    tdm_load_2d(XhB, kOffA + 0 * kABuf, 16, 16);
    tdm_load_2d(XlB, kOffA + 1 * kABuf, 16, 16);
  }

  float runmin = 3.4e38f;
  const unsigned aLane = (unsigned)(ml * kSK + 8 * hh) * 2;   // byte offset within one A array
  const bf16_t* bRowH = (const bf16_t*)(smem + kOffBH) + (wave * 16 + ml) * kSK + 8 * hh;
  const bf16_t* bRowL = (const bf16_t*)(smem + kOffBL) + (wave * 16 + ml) * kSK + 8 * hh;

  for (int mt = 0; mt < 49; ++mt) {
    const unsigned buf = (unsigned)(mt & 1), nxt = buf ^ 1u;
    __syncthreads();  // everyone done computing tile mt-1 (which lived in `nxt`)
    if (wave == 0) {
      if (mt + 1 < 49) {
        const size_t off = (size_t)(mt + 1) * 16 * kC;
        tdm_load_2d(XhB + off, kOffA + (2 * nxt + 0) * kABuf, 16, 16);
        tdm_load_2d(XlB + off, kOffA + (2 * nxt + 1) * kABuf, 16, 16);
        __builtin_amdgcn_s_wait_tensorcnt(2);  // in-order: tile mt (and B) now in LDS
      } else {
        __builtin_amdgcn_s_wait_tensorcnt(0);
      }
    }
    __syncthreads();  // publish tile mt

    const bf16_t* aRowH = (const bf16_t*)(smem + (kOffA + (2 * buf + 0) * kABuf) + aLane);
    const bf16_t* aRowL = (const bf16_t*)(smem + (kOffA + (2 * buf + 1) * kABuf) + aLane);
    v8f acc[4];
#pragma unroll
    for (int a = 0; a < 4; ++a) acc[a] = (v8f){0.f,0.f,0.f,0.f,0.f,0.f,0.f,0.f};

#pragma unroll
    for (int k0 = 0; k0 < kC; k0 += 32) {          // fully unrolled: 16 chunks, 48 WMMAs
      v16bf aH = CAT16(*(const v8bf*)(aRowH + k0), *(const v8bf*)(aRowH + k0 + 16));
      v16bf aL = CAT16(*(const v8bf*)(aRowL + k0), *(const v8bf*)(aRowL + k0 + 16));
      v16bf bH = CAT16(*(const v8bf*)(bRowH + k0), *(const v8bf*)(bRowH + k0 + 16));
      v16bf bL = CAT16(*(const v8bf*)(bRowL + k0), *(const v8bf*)(bRowL + k0 + 16));
      const int a = (k0 >> 5) & 3;                  // constant after unroll: 4-deep ILP
      acc[a] = __builtin_amdgcn_wmma_f32_16x16x32_bf16(false, aH, false, bH, (short)0, acc[a], false, false);
      acc[a] = __builtin_amdgcn_wmma_f32_16x16x32_bf16(false, aH, false, bL, (short)0, acc[a], false, false);
      acc[a] = __builtin_amdgcn_wmma_f32_16x16x32_bf16(false, aL, false, bH, (short)0, acc[a], false, false);
    }

    const float* x2p = x2w + b * kHW + mt * 16 + 8 * hh;
    float4 xq0 = *(const float4*)x2p;
    float4 xq1 = *(const float4*)(x2p + 4);
    float xr[8] = {xq0.x, xq0.y, xq0.z, xq0.w, xq1.x, xq1.y, xq1.z, xq1.w};
#pragma unroll
    for (int r = 0; r < 8; ++r) {
      float xp = (acc[0][r] + acc[1][r]) + (acc[2][r] + acc[3][r]);
      float d2 = fmaf(-2.f, xp, xr[r] + p2v);
      runmin = fminf(runmin, __builtin_sqrtf(fmaxf(d2, 0.f)));
    }
  }

  runmin = fminf(runmin, __shfl_xor(runmin, 16, 32));
  if (hh == 0 && gN < kP) out[(size_t)b * kP + gN] = -runmin;
}

// ============ fallback (small d_ws): in-kernel conversion path ============
__global__ __launch_bounds__(128) void proto_fb(const float* __restrict__ feat,
                                                const float* __restrict__ prot,
                                                const float* __restrict__ x2w,
                                                const float* __restrict__ p2w,
                                                float* __restrict__ out) {
  extern __shared__ char smem[];
  bf16_t* Bh = (bf16_t*)smem;
  bf16_t* Bl = Bh + kNT * kSK;
  bf16_t* Ah = Bl + kNT * kSK;
  bf16_t* Al = Ah + 16 * kSKA;

  const int tid = threadIdx.x, lane = tid & 31, wave = tid >> 5;
  const int ml = lane & 15, hh = lane >> 4;
  const int b = blockIdx.x >> 5, n0 = (blockIdx.x & 31) * kNT;
  const int gN = n0 + wave * 16 + ml;
  const float p2v = (gN < kP) ? p2w[gN] : 0.f;
  const float* featB = feat + (size_t)b * kC * kHW;

  for (int idx = tid; idx < kNT * (kC / 8); idx += 128) {
    int n = idx >> 6, c8 = (idx & 63) << 3, gn = n0 + n;
    float va[8];
    if (gn < kP) {
      const float4* src = (const float4*)(prot + (size_t)gn * kC + c8);
      float4 q0 = src[0], q1 = src[1];
      va[0]=q0.x; va[1]=q0.y; va[2]=q0.z; va[3]=q0.w;
      va[4]=q1.x; va[5]=q1.y; va[6]=q1.z; va[7]=q1.w;
    } else {
#pragma unroll
      for (int i = 0; i < 8; ++i) va[i] = 0.f;
    }
    v8bf hi, lo;
#pragma unroll
    for (int i = 0; i < 8; ++i) {
      bf16_t h = (bf16_t)va[i];
      hi[i] = h; lo[i] = (bf16_t)(va[i] - (float)h);
    }
    *(v8bf*)(Bh + n * kSK + c8) = hi;
    *(v8bf*)(Bl + n * kSK + c8) = lo;
  }
  __syncthreads();

  const bf16_t* aRowH = Ah + ml * kSKA + 8 * hh;
  const bf16_t* aRowL = Al + ml * kSKA + 8 * hh;
  const bf16_t* bRowH = Bh + (wave * 16 + ml) * kSK + 8 * hh;
  const bf16_t* bRowL = Bl + (wave * 16 + ml) * kSK + 8 * hh;
  float runmin = 3.4e38f;

  for (int m0 = 0; m0 < kHW; m0 += 16) {
    v8f acc0 = {0.f,0.f,0.f,0.f,0.f,0.f,0.f,0.f};
    v8f acc1 = {0.f,0.f,0.f,0.f,0.f,0.f,0.f,0.f};
    for (int kc = 0; kc < kC; kc += kKC) {
      __syncthreads();
      {
        const int c0 = kc + (tid << 1);
        const float4* s0 = (const float4*)(featB + (size_t)c0 * kHW + m0);
        const float4* s1 = (const float4*)(featB + (size_t)(c0 + 1) * kHW + m0);
        float ua[16], wa[16];
#pragma unroll
        for (int q = 0; q < 4; ++q) {
          float4 t0 = s0[q], t1 = s1[q];
          ua[4*q+0]=t0.x; ua[4*q+1]=t0.y; ua[4*q+2]=t0.z; ua[4*q+3]=t0.w;
          wa[4*q+0]=t1.x; wa[4*q+1]=t1.y; wa[4*q+2]=t1.z; wa[4*q+3]=t1.w;
        }
        const int col = tid << 1;
#pragma unroll
        for (int m = 0; m < 16; ++m) {
          bf16_t h0 = (bf16_t)ua[m], h1 = (bf16_t)wa[m];
          v2bf hp; hp[0] = h0; hp[1] = h1;
          *(v2bf*)(Ah + m * kSKA + col) = hp;
          v2bf lp;
          lp[0] = (bf16_t)(ua[m] - (float)h0);
          lp[1] = (bf16_t)(wa[m] - (float)h1);
          *(v2bf*)(Al + m * kSKA + col) = lp;
        }
      }
      __syncthreads();
#pragma unroll
      for (int k0 = 0; k0 < kKC; k0 += 32) {
        v16bf aH = CAT16(*(const v8bf*)(aRowH + k0), *(const v8bf*)(aRowH + k0 + 16));
        v16bf aL = CAT16(*(const v8bf*)(aRowL + k0), *(const v8bf*)(aRowL + k0 + 16));
        const int kg = kc + k0;
        v16bf bH = CAT16(*(const v8bf*)(bRowH + kg), *(const v8bf*)(bRowH + kg + 16));
        v16bf bL = CAT16(*(const v8bf*)(bRowL + kg), *(const v8bf*)(bRowL + kg + 16));
        if ((k0 >> 5) & 1) {
          acc1 = __builtin_amdgcn_wmma_f32_16x16x32_bf16(false, aH, false, bH, (short)0, acc1, false, false);
          acc1 = __builtin_amdgcn_wmma_f32_16x16x32_bf16(false, aH, false, bL, (short)0, acc1, false, false);
          acc1 = __builtin_amdgcn_wmma_f32_16x16x32_bf16(false, aL, false, bH, (short)0, acc1, false, false);
        } else {
          acc0 = __builtin_amdgcn_wmma_f32_16x16x32_bf16(false, aH, false, bH, (short)0, acc0, false, false);
          acc0 = __builtin_amdgcn_wmma_f32_16x16x32_bf16(false, aH, false, bL, (short)0, acc0, false, false);
          acc0 = __builtin_amdgcn_wmma_f32_16x16x32_bf16(false, aL, false, bH, (short)0, acc0, false, false);
        }
      }
    }
    const float* x2p = x2w + b * kHW + m0 + 8 * hh;
    float4 xq0 = *(const float4*)x2p;
    float4 xq1 = *(const float4*)(x2p + 4);
    float xr[8] = {xq0.x, xq0.y, xq0.z, xq0.w, xq1.x, xq1.y, xq1.z, xq1.w};
#pragma unroll
    for (int r = 0; r < 8; ++r) {
      float d2 = fmaf(-2.f, acc0[r] + acc1[r], xr[r] + p2v);
      runmin = fminf(runmin, __builtin_sqrtf(fmaxf(d2, 0.f)));
    }
  }
  runmin = fminf(runmin, __shfl_xor(runmin, 16, 32));
  if (hh == 0 && gN < kP) out[(size_t)b * kP + gN] = -runmin;
}

extern "C" void kernel_launch(void* const* d_in, const int* in_sizes, int n_in,
                              void* d_out, int out_size, void* d_ws, size_t ws_size,
                              hipStream_t stream) {
  const float* feat = (const float*)d_in[0];
  const float* prot = (const float*)d_in[1];
  float* out = (float*)d_out;
  char* ws = (char*)d_ws;
  float* x2w = (float*)(ws + kOffX2);
  float* p2w = (float*)(ws + kOffP2);

  x2_kernel<<<kB * 4, 256, 0, stream>>>(feat, x2w);
  p2_kernel<<<kP, 32, 0, stream>>>(prot, p2w);

  if (ws_size >= kWsTdm) {
    bf16_t* Xh = (bf16_t*)(ws + kOffXh);
    bf16_t* Xl = (bf16_t*)(ws + kOffXl);
    bf16_t* Ph = (bf16_t*)(ws + kOffPh);
    bf16_t* Pl = (bf16_t*)(ws + kOffPl);
    cvt_feat<<<dim3(25, 16, kB), 256, 0, stream>>>(feat, Xh, Xl);
    cvt_prot<<<(kP * kC + 255) / 256, 256, 0, stream>>>(prot, Ph, Pl);
    proto_tdm<<<kB * 32, 128, kSmemTdm, stream>>>(Xh, Xl, Ph, Pl, x2w, p2w, out);
  } else {
    proto_fb<<<kB * 32, 128, kSmemFb, stream>>>(feat, prot, x2w, p2w, out);
  }
}
